// GAT_Model_27075473834576
// MI455X (gfx1250) — matrix operand order repaired and checked by reading, not compiled, not run on hardware
//
#include <hip/hip_runtime.h>
#include <hip/hip_bf16.h>
#include <hip/hip_fp16.h>

// ---------------- problem constants (match reference) ----------------
#define N_NODES 50000
#define N_EDGES 800000
#define FIN     128
#define DHEAD   64
#define H       3
#define HD      192           // H * DHEAD
#define PRED_K  384           // 2 * HD
#define PRED_N  16            // CLS padded up to one WMMA tile
#define CLS     10
#define NEG_SLOPE 0.2f

typedef __attribute__((ext_vector_type(16))) _Float16 v16h;
typedef __attribute__((ext_vector_type(8)))  _Float16 v8h;
typedef __attribute__((ext_vector_type(8)))  float    v8f;

// ---------------- small utility kernels ----------------
__global__ void cvt_f32_to_f16(const float* __restrict__ s, _Float16* __restrict__ d, long n) {
    long t = (long)blockIdx.x * blockDim.x + threadIdx.x;
    if (t < n) d[t] = (_Float16)s[t];
}

__global__ void fill_f32(float* __restrict__ p, float v, long n) {
    long t = (long)blockIdx.x * blockDim.x + threadIdx.x;
    if (t < n) p[t] = v;
}

// Pack a row-major f32 weight [K, Nc] into f16 "B-fragment order":
//   packed[(kt*Nc + c)*32 + g*16 + i] = W[(kt*32 + g*16 + i)*Nc + c]
// so each lane's 16 B halves are two 16-byte runs, at 8*(l>>4) and 16+8*(l>>4) in the 32.
__global__ void pack_b_f16(const float* __restrict__ W, _Float16* __restrict__ P,
                           int K, int Nc) {
    int t = blockIdx.x * blockDim.x + threadIdx.x;
    if (t >= K * Nc) return;
    int k = t / Nc, c = t % Nc;
    size_t dst = (size_t)((k >> 5) * Nc + c) * 32 + (((k >> 4) & 1) * 16) + (k & 15);
    P[dst] = (_Float16)W[t];
}

// Same, for Wp [K, CLS] zero-padded to Nc=PRED_N columns.
__global__ void pack_wp_f16(const float* __restrict__ Wp, _Float16* __restrict__ P) {
    int t = blockIdx.x * blockDim.x + threadIdx.x;
    if (t >= PRED_K * PRED_N) return;
    int k = t / PRED_N, c = t % PRED_N;
    size_t dst = (size_t)((k >> 5) * PRED_N + c) * 32 + (((k >> 4) & 1) * 16) + (k & 15);
    P[dst] = (_Float16)((c < CLS) ? Wp[k * CLS + c] : 0.0f);
}

// float atomic max via int/uint punning (valid with -FLT_MAX-ish init)
__device__ inline void atomicMaxFloat(float* addr, float val) {
    if (val >= 0.0f) atomicMax((int*)addr, __float_as_int(val));
    else             atomicMin((unsigned int*)addr, __float_as_uint(val));
}

// ---- fragment helpers (CDNA5 16x16x32 f16 layouts, wave32) ----
// A (16x32): lane l holds row l&15; halves 0..7 -> K=kb..kb+7, halves 8..15 -> K=16+kb..+7,
// kb = 8*(l>>4). Two aligned 16B loads.
__device__ inline v16h load_a_frag(const _Float16* rowp, int k0, int kb) {
    const v8h a0 = *(const v8h*)(rowp + k0 + kb);
    const v8h a1 = *(const v8h*)(rowp + k0 + 16 + kb);
    v16h a;
#pragma unroll
    for (int i = 0; i < 8; ++i) { a[i] = a0[i]; a[8 + i] = a1[i]; }
    return a;
}

// B fragment from packed buffer: two aligned 16B loads per lane.
__device__ inline v16h load_b_frag_packed(const _Float16* __restrict__ P, int kt,
                                          int colbase, int lane, int Nc) {
    const _Float16* p = P + (size_t)(kt * Nc + colbase + (lane & 15)) * 32 + ((lane >> 4) * 8);
    const v8h b0 = *(const v8h*)p;
    const v8h b1 = *(const v8h*)(p + 16);
    v16h b;
#pragma unroll
    for (int i = 0; i < 8; ++i) { b[i] = b0[i]; b[8 + i] = b1[i]; }
    return b;
}

// ---------------- WMMA GEMM: C[M,Nc] = A[M,K] * Bpacked ----------------
// One wave computes MT vertically-adjacent 16x16 tiles. Per k-step: load B frag (2xB128),
// preload all MT A frags (distinct regs, clause-batched), then MT back-to-back WMMAs
// (no VALU writes to WMMA sources between them -> no WAR hazard NOPs).
template <int MT, int K, int Nc>
__global__ __launch_bounds__(32)
void gemm_wmma_f16(const _Float16* __restrict__ A, const _Float16* __restrict__ Bp,
                   float* __restrict__ C) {
    const int lane = threadIdx.x;                   // 0..31 (wave32)
    const int colbase = blockIdx.y * 16;
    const int col  = colbase + (lane & 15);
    const int kb   = (lane >> 4) * 8;
    const int row0 = blockIdx.x * (16 * MT) + (lane & 15);

    const _Float16* arow[MT];
#pragma unroll
    for (int m = 0; m < MT; ++m) arow[m] = A + (size_t)(row0 + m * 16) * K;

    v8f acc[MT];
#pragma unroll
    for (int m = 0; m < MT; ++m) acc[m] = (v8f){};

#pragma unroll
    for (int k0 = 0; k0 < K; k0 += 32) {
        const v16h b = load_b_frag_packed(Bp, k0 >> 5, colbase, lane, Nc);
        v16h a[MT];
#pragma unroll
        for (int m = 0; m < MT; ++m) a[m] = load_a_frag(arow[m], k0, kb);
#pragma unroll
        for (int m = 0; m < MT; ++m)
            acc[m] = __builtin_amdgcn_wmma_f32_16x16x32_f16(false, a[m], false, b,
                                                            (short)0, acc[m], false, false);
    }
    // store: fragment value i lives at row (tile*16 + i + 8*(lane>>4)), col (lane&15)
    const int rsub = (lane >> 4) << 3;
#pragma unroll
    for (int m = 0; m < MT; ++m) {
        const int rbase = blockIdx.x * (16 * MT) + m * 16 + rsub;
#pragma unroll
        for (int i = 0; i < 8; ++i)
            C[(size_t)(rbase + i) * Nc + col] = acc[m][i];
    }
}

// ---------------- attention logits: el/er [N,H] ----------------
__global__ void el_er_kernel(const float* __restrict__ f, const float* __restrict__ al,
                             const float* __restrict__ ar, float* __restrict__ el,
                             float* __restrict__ er) {
    int t = blockIdx.x * blockDim.x + threadIdx.x;
    if (t >= N_NODES * H) return;
    int node = t / H, h = t % H;
    const float* fp  = f  + (size_t)node * HD + h * DHEAD;
    const float* alp = al + h * DHEAD;
    const float* arp = ar + h * DHEAD;
    float sl = 0.0f, sr = 0.0f;
#pragma unroll 4
    for (int i = 0; i < DHEAD; ++i) { float v = fp[i]; sl += v * alp[i]; sr += v * arp[i]; }
    el[t] = sl; er[t] = sr;
}

// ---------------- edge softmax pass 1: leaky_relu + segment max ----------------
__global__ void edge_logits_max(const int* __restrict__ src, const int* __restrict__ dst,
                                const float* __restrict__ el, const float* __restrict__ er,
                                float* __restrict__ ev, float* __restrict__ mmax) {
    int e = blockIdx.x * blockDim.x + threadIdx.x;
    if (e >= N_EDGES) return;
    int s = src[e], d = dst[e];
#pragma unroll
    for (int h = 0; h < H; ++h) {
        float x = el[s * H + h] + er[d * H + h];
        x = (x > 0.0f) ? x : NEG_SLOPE * x;
        ev[(size_t)e * H + h] = x;
        atomicMaxFloat(&mmax[d * H + h], x);
    }
}

// ---------------- edge softmax pass 2: exp + segment sum ----------------
__global__ void edge_exp_sum(const int* __restrict__ dst, const float* __restrict__ ev,
                             const float* __restrict__ mmax, float* __restrict__ exv,
                             float* __restrict__ denom) {
    int e = blockIdx.x * blockDim.x + threadIdx.x;
    if (e >= N_EDGES) return;
    int d = dst[e];
#pragma unroll
    for (int h = 0; h < H; ++h) {
        float x = __expf(ev[(size_t)e * H + h] - mmax[d * H + h]);
        exv[(size_t)e * H + h] = x;
        atomicAdd(&denom[d * H + h], x);
    }
}

// ---------------- edge softmax pass 3: weighted aggregation ----------------
__global__ void edge_aggregate(const int* __restrict__ src, const int* __restrict__ dst,
                               const float* __restrict__ f, const float* __restrict__ exv,
                               const float* __restrict__ denom, float* __restrict__ agg) {
    long t = (long)blockIdx.x * blockDim.x + threadIdx.x;
    if (t >= (long)N_EDGES * DHEAD) return;
    int e  = (int)(t / DHEAD);
    int dd = (int)(t % DHEAD);
    int s = src[e], d = dst[e];
#pragma unroll
    for (int h = 0; h < H; ++h) {
        float alpha = exv[(size_t)e * H + h] / denom[d * H + h];
        atomicAdd(&agg[(size_t)d * HD + h * DHEAD + dd],
                  f[(size_t)s * HD + h * DHEAD + dd] * alpha);
    }
}

// ---------------- bias + relu + f16 downconvert ----------------
__global__ void bias_relu_f16(const float* __restrict__ agg, const float* __restrict__ b,
                              _Float16* __restrict__ hh) {
    long t = (long)blockIdx.x * blockDim.x + threadIdx.x;
    if (t >= (long)N_NODES * HD) return;
    int c = (int)(t % HD);
    float v = agg[t] + b[c];
    hh[t] = (_Float16)((v > 0.0f) ? v : 0.0f);
}

// ---------------- predictor: WMMA GEMM with per-edge concat-gather ----------------
// A row e = [h2[src[e]], h2[dst[e]]] (f16, K=384); B = packed Wp [384,16]; out f32 [E,10].
template <int MT>
__global__ __launch_bounds__(32)
void pred_wmma(const int* __restrict__ src, const int* __restrict__ dst,
               const _Float16* __restrict__ Hf, const _Float16* __restrict__ Wpp,
               const float* __restrict__ bp, float* __restrict__ out) {
    const int lane = threadIdx.x;
    const int col  = lane & 15;
    const int kb   = (lane >> 4) * 8;

    const _Float16* hs[MT];
    const _Float16* hd[MT];
#pragma unroll
    for (int m = 0; m < MT; ++m) {
        const int e = (blockIdx.x * MT + m) * 16 + (lane & 15);
        hs[m] = Hf + (size_t)src[e] * HD;
        hd[m] = Hf + (size_t)dst[e] * HD;
    }

    v8f acc[MT];
#pragma unroll
    for (int m = 0; m < MT; ++m) acc[m] = (v8f){};

#pragma unroll
    for (int k0 = 0; k0 < PRED_K; k0 += 32) {
        const v16h b = load_b_frag_packed(Wpp, k0 >> 5, 0, lane, PRED_N);
        v16h a[MT];
#pragma unroll
        for (int m = 0; m < MT; ++m) {
            // 8-half runs never straddle the src/dst boundary (HD % 32 == 0);
            // side selection depends only on k0 -> folds at unroll time.
            const _Float16* p0 = (k0 < HD)      ? hs[m] + k0 + kb        : hd[m] + (k0 - HD) + kb;
            const _Float16* p1 = (k0 + 16 < HD) ? hs[m] + k0 + 16 + kb   : hd[m] + (k0 + 16 - HD) + kb;
            const v8h a0 = *(const v8h*)p0;
            const v8h a1 = *(const v8h*)p1;
#pragma unroll
            for (int i = 0; i < 8; ++i) { a[m][i] = a0[i]; a[m][8 + i] = a1[i]; }
        }
#pragma unroll
        for (int m = 0; m < MT; ++m)
            acc[m] = __builtin_amdgcn_wmma_f32_16x16x32_f16(false, a[m], false, b,
                                                            (short)0, acc[m], false, false);
    }

    if (col < CLS) {
        const float bias = bp[col];
        const int rsub = (lane >> 4) << 3;
#pragma unroll
        for (int m = 0; m < MT; ++m) {
            const int ebase = (blockIdx.x * MT + m) * 16 + rsub;
#pragma unroll
            for (int i = 0; i < 8; ++i)
                out[(size_t)(ebase + i) * CLS + col] = acc[m][i] + bias;
        }
    }
}

// ---------------- host orchestration ----------------
extern "C" void kernel_launch(void* const* d_in, const int* in_sizes, int n_in,
                              void* d_out, int out_size, void* d_ws, size_t ws_size,
                              hipStream_t stream) {
    (void)in_sizes; (void)n_in; (void)out_size; (void)ws_size;
    const int*   src    = (const int*)  d_in[0];
    const int*   dst    = (const int*)  d_in[1];
    const float* nfeats = (const float*)d_in[2];
    /* d_in[3] = efeats, unused by reference */
    const float* W1  = (const float*)d_in[4];
    const float* al1 = (const float*)d_in[5];
    const float* ar1 = (const float*)d_in[6];
    const float* b1  = (const float*)d_in[7];
    const float* W2  = (const float*)d_in[8];
    const float* al2 = (const float*)d_in[9];
    const float* ar2 = (const float*)d_in[10];
    const float* b2  = (const float*)d_in[11];
    const float* Wp  = (const float*)d_in[12];
    const float* bp  = (const float*)d_in[13];
    float* out = (float*)d_out;

    // carve workspace
    char* w = (char*)d_ws;
    auto carve = [&](size_t bytes) -> char* {
        char* p = w; w += (bytes + 255) & ~(size_t)255; return p;
    };
    _Float16* nf16  = (_Float16*)carve((size_t)N_NODES * FIN * 2);
    _Float16* w1p   = (_Float16*)carve((size_t)FIN * HD * 2);
    _Float16* w2p   = (_Float16*)carve((size_t)HD * HD * 2);
    _Float16* wpp   = (_Float16*)carve((size_t)PRED_K * PRED_N * 2);
    float*    fproj = (float*)   carve((size_t)N_NODES * HD * 4);
    float*    elv   = (float*)   carve((size_t)N_NODES * H * 4);
    float*    erv   = (float*)   carve((size_t)N_NODES * H * 4);
    float*    mmax  = (float*)   carve((size_t)N_NODES * H * 4);
    float*    denom = (float*)   carve((size_t)N_NODES * H * 4);
    float*    exv   = (float*)   carve((size_t)N_EDGES * H * 4);
    float*    agg   = (float*)   carve((size_t)N_NODES * HD * 4);
    _Float16* h1h   = (_Float16*)carve((size_t)N_NODES * HD * 2);
    _Float16* h2h   = (_Float16*)carve((size_t)N_NODES * HD * 2);

    const int T = 256;
    auto blocks = [](long n, int t) { return (unsigned)((n + t - 1) / t); };

    // ---- convert node features; pack weights into B-fragment order ----
    cvt_f32_to_f16<<<blocks((long)N_NODES * FIN, T), T, 0, stream>>>(nfeats, nf16, (long)N_NODES * FIN);
    pack_b_f16<<<blocks((long)FIN * HD, T), T, 0, stream>>>(W1, w1p, FIN, HD);
    pack_b_f16<<<blocks((long)HD * HD, T), T, 0, stream>>>(W2, w2p, HD, HD);
    pack_wp_f16<<<blocks((long)PRED_K * PRED_N, T), T, 0, stream>>>(Wp, wpp);

    const long nh  = (long)N_NODES * H;
    const long nhd = (long)N_NODES * HD;
    const long ed  = (long)N_EDGES * DHEAD;

    // 50000/16 = 3125 row tiles = 625 blocks of MT=5
    constexpr int GEMM_MT = 5;
    const dim3 gemm_grid(N_NODES / (16 * GEMM_MT), HD / 16);
    // predictor: 800000/16 = 50000 edge tiles = 12500 blocks of MT=4
    constexpr int PRED_MT = 4;
    const unsigned pred_grid = N_EDGES / (16 * PRED_MT);

    // =============== Layer 1 ===============
    gemm_wmma_f16<GEMM_MT, FIN, HD><<<gemm_grid, 32, 0, stream>>>(nf16, w1p, fproj);
    el_er_kernel<<<blocks(nh, T), T, 0, stream>>>(fproj, al1, ar1, elv, erv);
    fill_f32<<<blocks(nh, T), T, 0, stream>>>(mmax, -3.0e38f, nh);
    fill_f32<<<blocks(nh, T), T, 0, stream>>>(denom, 0.0f, nh);
    fill_f32<<<blocks(nhd, T), T, 0, stream>>>(agg, 0.0f, nhd);
    edge_logits_max<<<blocks(N_EDGES, T), T, 0, stream>>>(src, dst, elv, erv, exv, mmax);
    edge_exp_sum<<<blocks(N_EDGES, T), T, 0, stream>>>(dst, exv, mmax, exv, denom);
    edge_aggregate<<<blocks(ed, T), T, 0, stream>>>(src, dst, fproj, exv, denom, agg);
    bias_relu_f16<<<blocks(nhd, T), T, 0, stream>>>(agg, b1, h1h);

    // =============== Layer 2 ===============
    gemm_wmma_f16<GEMM_MT, HD, HD><<<gemm_grid, 32, 0, stream>>>(h1h, w2p, fproj);
    el_er_kernel<<<blocks(nh, T), T, 0, stream>>>(fproj, al2, ar2, elv, erv);
    fill_f32<<<blocks(nh, T), T, 0, stream>>>(mmax, -3.0e38f, nh);
    fill_f32<<<blocks(nh, T), T, 0, stream>>>(denom, 0.0f, nh);
    fill_f32<<<blocks(nhd, T), T, 0, stream>>>(agg, 0.0f, nhd);
    edge_logits_max<<<blocks(N_EDGES, T), T, 0, stream>>>(src, dst, elv, erv, exv, mmax);
    edge_exp_sum<<<blocks(N_EDGES, T), T, 0, stream>>>(dst, exv, mmax, exv, denom);
    edge_aggregate<<<blocks(ed, T), T, 0, stream>>>(src, dst, fproj, exv, denom, agg);
    bias_relu_f16<<<blocks(nhd, T), T, 0, stream>>>(agg, b2, h2h);

    // =============== Edge predictor ===============
    pred_wmma<PRED_MT><<<pred_grid, 32, 0, stream>>>(src, dst, h2h, wpp, bp, out);
}
